// TransformerFeatureExtractor_44238163149420
// MI455X (gfx1250) — matrix operand order repaired and checked by reading, not compiled, not run on hardware
//
#include <hip/hip_runtime.h>
#include <cmath>

// ---------------- WMMA types (CDNA5 / gfx1250, wave32) ----------------
typedef __attribute__((ext_vector_type(16))) __bf16 v16bf;
typedef __attribute__((ext_vector_type(8)))  float  v8f;

#define BM 64
#define BN 64
#define BK 32
#define LDW 20   // LDS row stride in dwords (16 data + 4 pad, 16B aligned)

// Model dims
#define B_   16
#define S_   1024
#define IND_ 64
#define D_   512
#define H_   8
#define HD_  64
#define FF_  2048
#define L_   6
#define OUT_ 256
#define BS_  (B_ * S_)

// fp32 <-> bf16 helpers
__device__ __forceinline__ uint16_t bf16_rne(float x) {
  uint32_t u = __float_as_uint(x);
  u += 0x7FFFu + ((u >> 16) & 1u);
  return (uint16_t)(u >> 16);
}
__device__ __forceinline__ float bf16_to_f(uint16_t h) {
  return __uint_as_float(((uint32_t)h) << 16);
}
__device__ __forceinline__ uint32_t pack_bf16(float a, float b) {
  uint32_t ua = __float_as_uint(a);
  uint32_t ub = __float_as_uint(b);
  ua += 0x7FFFu + ((ua >> 16) & 1u);
  ub += 0x7FFFu + ((ub >> 16) & 1u);
  return (ua >> 16) | (ub & 0xFFFF0000u);
}

// ------------------- fp32 -> bf16 bulk convert (weights, x) -----------------
__global__ __launch_bounds__(256) void cvt_bf16_kernel(
    const float* __restrict__ in, uint16_t* __restrict__ out, long n4) {
  long i = (long)blockIdx.x * 256 + threadIdx.x;
  if (i < n4) {
    float4 v = ((const float4*)in)[i];
    uint2 o;
    o.x = pack_bf16(v.x, v.y);
    o.y = pack_bf16(v.z, v.w);
    ((uint2*)out)[i] = o;
  }
}

// ---------------------------------------------------------------------
// Batched GEMM on bf16 operands: C = scale * (A @ B[^T]) + bias [, GELU].
// Two-level batch: z -> outer zo = z/zdiv (strides *o), inner zi = z%zdiv
// (strides *i); lets one launch cover all (batch, head) pairs.
// 128 threads = 4 waves; each wave owns a 32x32 tile (2x2 WMMA frags),
// fp32 accumulate, ping-pong LDS, one barrier per K-step.
// ---------------------------------------------------------------------
template <bool BT, bool GELU, bool OUTBF>
__global__ __launch_bounds__(128) void gemm_bf16_wmma(
    const uint16_t* __restrict__ A, const uint16_t* __restrict__ Bm,
    const float* __restrict__ bias, void* __restrict__ Cv,
    int M, int N, int K, int lda, int ldb, int ldc, int zdiv,
    long sAi, long sAo, long sBi, long sBo, long sCi, long sCo, float scale) {
  __shared__ uint32_t As[2][BM][LDW];  // [buf][row][k-pair dword]
  __shared__ uint32_t Bs[2][BN][LDW];  // [buf][col][k-pair dword]

  const int tid  = threadIdx.x;
  const int lane = tid & 31;
  const int wave = tid >> 5;
  const int l16  = lane & 15;
  const int half = lane >> 4;

  const int mblk = blockIdx.y * BM;
  const int nblk = blockIdx.x * BN;
  const int zo = blockIdx.z / zdiv;
  const int zi = blockIdx.z - zo * zdiv;
  A  += (long)zo * sAo + (long)zi * sAi;
  Bm += (long)zo * sBo + (long)zi * sBi;
  const long coff = (long)zo * sCo + (long)zi * sCi;

  const int wm = (wave >> 1) * 32;
  const int wn = (wave & 1) * 32;

  v8f acc[2][2] = {};

  uint4 ra[2];   // staged A: 2 x 16B (8 bf16 each)
  uint4 rb[2];   // staged B

  auto load_tile = [&](int k0) {
#pragma unroll
    for (int j = 0; j < 2; ++j) {
      int e  = tid + j * 128;       // 0..255
      int r  = e & 63;              // row in tile
      int d4 = (e >> 6) * 4;        // dword offset in row: 0/4/8/12
      int gr = mblk + r;
      ra[j] = uint4{0u, 0u, 0u, 0u};
      if (gr < M) ra[j] = *(const uint4*)(A + (long)gr * lda + k0 + d4 * 2);
    }
    if (BT) {  // B is N x K bf16 row-major: straight copy
#pragma unroll
      for (int j = 0; j < 2; ++j) {
        int e  = tid + j * 128;
        int r  = e & 63;
        int d4 = (e >> 6) * 4;
        int gn = nblk + r;
        rb[j] = uint4{0u, 0u, 0u, 0u};
        if (gn < N) rb[j] = *(const uint4*)(Bm + (long)gn * ldb + k0 + d4 * 2);
      }
    } else {   // B is K x N bf16: two adjacent k rows, 4 cols each
#pragma unroll
      for (int j = 0; j < 2; ++j) {
        int e  = tid + j * 128;     // 0..255
        int kk = e >> 4;            // k-pair 0..15
        int n4 = (e & 15) * 4;
        int gk = k0 + 2 * kk;
        int gn = nblk + n4;
        uint2 lo = uint2{0u, 0u}, hi = uint2{0u, 0u};
        if (gn < N) {
          lo = *(const uint2*)(Bm + (long)gk * ldb + gn);
          hi = *(const uint2*)(Bm + (long)(gk + 1) * ldb + gn);
        }
        rb[j] = uint4{lo.x, lo.y, hi.x, hi.y};
      }
    }
  };

  auto store_tile = [&](int buf) {
#pragma unroll
    for (int j = 0; j < 2; ++j) {
      int e  = tid + j * 128;
      int r  = e & 63;
      int d4 = (e >> 6) * 4;
      *(uint4*)&As[buf][r][d4] = ra[j];
    }
    if (BT) {
#pragma unroll
      for (int j = 0; j < 2; ++j) {
        int e  = tid + j * 128;
        int r  = e & 63;
        int d4 = (e >> 6) * 4;
        *(uint4*)&Bs[buf][r][d4] = rb[j];
      }
    } else {
#pragma unroll
      for (int j = 0; j < 2; ++j) {
        int e  = tid + j * 128;
        int kk = e >> 4;
        int n4 = (e & 15) * 4;
        uint32_t lx = rb[j].x, ly = rb[j].y, hx = rb[j].z, hy = rb[j].w;
        Bs[buf][n4 + 0][kk] = (hx << 16) | (lx & 0xFFFFu);
        Bs[buf][n4 + 1][kk] = (hx & 0xFFFF0000u) | (lx >> 16);
        Bs[buf][n4 + 2][kk] = (hy << 16) | (ly & 0xFFFFu);
        Bs[buf][n4 + 3][kk] = (hy & 0xFFFF0000u) | (ly >> 16);
      }
    }
  };

  const int nk = K / BK;  // all call sites use K % 32 == 0
  load_tile(0);
  store_tile(0);
  __syncthreads();

  for (int t = 0; t < nk; ++t) {
    const int cur = t & 1;
    const bool has_next = (t + 1) < nk;
    if (has_next) load_tile((t + 1) * BK);  // global loads fly over WMMAs

    union Frag { uint32_t u[8]; v16bf v; };
    Frag fa[2], fb[2];
#pragma unroll
    for (int mi = 0; mi < 2; ++mi) {
      int row = wm + mi * 16 + l16;
#pragma unroll
      for (int q = 0; q < 8; ++q) {
        int p = (q & 3) + 4 * half + 8 * (q >> 2);  // A k-pair index
        fa[mi].u[q] = As[cur][row][p];
      }
    }
#pragma unroll
    for (int ni = 0; ni < 2; ++ni) {
      int col = wn + ni * 16 + l16;
#pragma unroll
      for (int q = 0; q < 8; ++q) {
        int p = (q & 3) + 4 * half + 8 * (q >> 2);  // B k-pair index
        fb[ni].u[q] = Bs[cur][col][p];
      }
    }
#pragma unroll
    for (int mi = 0; mi < 2; ++mi)
#pragma unroll
      for (int ni = 0; ni < 2; ++ni)
        acc[mi][ni] = __builtin_amdgcn_wmma_f32_16x16x32_bf16(
            false, fa[mi].v, false, fb[ni].v, (short)0, acc[mi][ni],
            false, false);

    if (has_next) {
      store_tile(cur ^ 1);
      __syncthreads();
    }
  }

  // ---- epilogue: scale, bias, optional exact GELU, fp32/bf16 store ----
  float*    Cf = (float*)Cv + coff;
  uint16_t* Ch = (uint16_t*)Cv + coff;
#pragma unroll
  for (int mi = 0; mi < 2; ++mi) {
#pragma unroll
    for (int ni = 0; ni < 2; ++ni) {
      int col = nblk + wn + ni * 16 + l16;
      if (col >= N) continue;
      float bv = bias ? bias[col] : 0.f;
#pragma unroll
      for (int r = 0; r < 8; ++r) {
        int row = mblk + wm + mi * 16 + r + 8 * half;
        if (row >= M) continue;
        float xv = acc[mi][ni][r] * scale + bv;
        if (GELU) xv = 0.5f * xv * (1.f + erff(xv * 0.70710678118654752f));
        if (OUTBF) Ch[(long)row * ldc + col] = bf16_rne(xv);
        else       Cf[(long)row * ldc + col] = xv;
      }
    }
  }
}

// ---- LayerNorm: fp32 in (+optional fp32 residual, +pos after LN),
//      fp32 out (residual stream) + optional bf16 mirror (GEMM feed) ----
__global__ __launch_bounds__(256) void ln_kernel(
    const float* __restrict__ x, const float* __restrict__ res,
    const float* __restrict__ pos, const float* __restrict__ g,
    const float* __restrict__ bb, float* __restrict__ out,
    uint16_t* __restrict__ outb, int width, int S) {
  __shared__ float s1[256], s2[256];
  int row = blockIdx.x;
  int tid = threadIdx.x;
  float vals[2] = {0.f, 0.f};
  float sum = 0.f, sq = 0.f;
  int nv = 0;
  for (int c = tid; c < width; c += 256) {
    float v = x[(long)row * width + c];
    if (res) v += res[(long)row * width + c];
    vals[nv++] = v;
    sum += v;
    sq += v * v;
  }
  s1[tid] = sum; s2[tid] = sq;
  __syncthreads();
  for (int off = 128; off > 0; off >>= 1) {
    if (tid < off) { s1[tid] += s1[tid + off]; s2[tid] += s2[tid + off]; }
    __syncthreads();
  }
  float mean = s1[0] / width;
  float var  = s2[0] / width - mean * mean;
  float inv  = rsqrtf(var + 1e-5f);
  nv = 0;
  for (int c = tid; c < width; c += 256) {
    float v = (vals[nv++] - mean) * inv * g[c] + bb[c];
    if (pos) v += pos[(long)(row % S) * width + c];
    out[(long)row * width + c] = v;
    if (outb) outb[(long)row * width + c] = bf16_rne(v);
  }
}

// ---- softmax: bf16 logits in, bf16 probabilities out (in place) ----
__global__ __launch_bounds__(256) void softmax_kernel(uint16_t* __restrict__ x,
                                                      int width) {
  __shared__ float sred[256];
  int tid = threadIdx.x;
  uint16_t* xr = x + (long)blockIdx.x * width;
  float vals[4];
  float mx = -3.0e38f;
  int nv = 0;
  for (int c = tid; c < width; c += 256) {
    vals[nv] = bf16_to_f(xr[c]);
    mx = fmaxf(mx, vals[nv]);
    ++nv;
  }
  sred[tid] = mx;
  __syncthreads();
  for (int off = 128; off > 0; off >>= 1) {
    if (tid < off) sred[tid] = fmaxf(sred[tid], sred[tid + off]);
    __syncthreads();
  }
  mx = sred[0];
  __syncthreads();
  float sum = 0.f;
  nv = 0;
  for (int c = tid; c < width; c += 256) {
    vals[nv] = expf(vals[nv] - mx);
    sum += vals[nv];
    ++nv;
  }
  sred[tid] = sum;
  __syncthreads();
  for (int off = 128; off > 0; off >>= 1) {
    if (tid < off) sred[tid] += sred[tid + off];
    __syncthreads();
  }
  float inv = 1.f / sred[0];
  nv = 0;
  for (int c = tid; c < width; c += 256) xr[c] = bf16_rne(vals[nv++] * inv);
}

// ---- mean pool over sequence: fp32 in, bf16 out (head GEMM feed) ----
__global__ __launch_bounds__(256) void pool_kernel(
    const float* __restrict__ h, uint16_t* __restrict__ pooled, int S, int Dd) {
  int b = blockIdx.x;
  for (int d = threadIdx.x; d < Dd; d += 256) {
    float s = 0.f;
    const float* p = h + (long)b * S * Dd + d;
    for (int t = 0; t < S; ++t) s += p[(long)t * Dd];
    pooled[b * Dd + d] = bf16_rne(s / (float)S);
  }
}

// ------------------------------- launchers ----------------------------------
static void launch_gemm(hipStream_t st, const uint16_t* A, const uint16_t* Bm,
                        const float* bias, void* C, int M, int N, int K,
                        int lda, int ldb, int ldc, int batch, int zdiv,
                        long sAi, long sAo, long sBi, long sBo, long sCi,
                        long sCo, float scale, bool bt, bool gelu, bool obf) {
  dim3 grid((N + BN - 1) / BN, (M + BM - 1) / BM, batch);
  dim3 blk(128);
#define GEMM_CALL(BT, GL, OB) \
  gemm_bf16_wmma<BT, GL, OB><<<grid, blk, 0, st>>>( \
      A, Bm, bias, C, M, N, K, lda, ldb, ldc, zdiv, sAi, sAo, sBi, sBo, \
      sCi, sCo, scale)
  if (bt)                GEMM_CALL(true,  false, true);   // scores (bf16 out)
  else if (gelu && obf)  GEMM_CALL(false, true,  true);   // ff1
  else if (gelu)         GEMM_CALL(false, true,  false);  // head1
  else if (obf)          GEMM_CALL(false, false, true);   // qkv, ctx
  else                   GEMM_CALL(false, false, false);  // fp32-out GEMMs
#undef GEMM_CALL
}

static void launch_cvt(hipStream_t st, const float* in, uint16_t* out, long n) {
  long n4 = n / 4;
  cvt_bf16_kernel<<<(n4 + 255) / 256, 256, 0, st>>>(in, out, n4);
}

extern "C" void kernel_launch(void* const* d_in, const int* in_sizes, int n_in,
                              void* d_out, int out_size, void* d_ws,
                              size_t ws_size, hipStream_t stream) {
  (void)in_sizes; (void)n_in; (void)out_size;
  const float* x      = (const float*)d_in[0];
  const float* in_w   = (const float*)d_in[1];
  const float* in_b   = (const float*)d_in[2];
  const float* in_lng = (const float*)d_in[3];
  const float* in_lnb = (const float*)d_in[4];
  const float* pos    = (const float*)d_in[5];
  const float* qkv_w  = (const float*)d_in[6];
  const float* qkv_b  = (const float*)d_in[7];
  const float* ao_w   = (const float*)d_in[8];
  const float* ao_b   = (const float*)d_in[9];
  const float* ff_w1  = (const float*)d_in[10];
  const float* ff_b1  = (const float*)d_in[11];
  const float* ff_w2  = (const float*)d_in[12];
  const float* ff_b2  = (const float*)d_in[13];
  const float* ln1_g  = (const float*)d_in[14];
  const float* ln1_b  = (const float*)d_in[15];
  const float* ln2_g  = (const float*)d_in[16];
  const float* ln2_b  = (const float*)d_in[17];
  const float* o_w1   = (const float*)d_in[18];
  const float* o_b1   = (const float*)d_in[19];
  const float* o_lng  = (const float*)d_in[20];
  const float* o_lnb  = (const float*)d_in[21];
  const float* o_w2   = (const float*)d_in[22];
  const float* o_b2   = (const float*)d_in[23];
  float* out = (float*)d_out;

  // ---- workspace partition ----
  char* wp = (char*)d_ws;
  auto allocf = [&](size_t n) { float* p = (float*)wp;    wp += n * 4; return p; };
  auto allocb = [&](size_t n) { uint16_t* p = (uint16_t*)wp; wp += n * 2; return p; };
  float*    hres  = allocf((size_t)BS_ * D_);       // fp32 residual stream
  float*    t0    = allocf((size_t)BS_ * D_);       // fp32 pre-LN GEMM out
  float*    z0    = allocf((size_t)B_ * (D_ / 2));
  float*    z1f   = allocf((size_t)B_ * (D_ / 2));
  uint16_t* xb    = allocb((size_t)BS_ * IND_);
  uint16_t* hb    = allocb((size_t)BS_ * D_);       // bf16 mirror of hres
  uint16_t* qkvb  = allocb((size_t)BS_ * 3 * D_);
  uint16_t* ctxb  = allocb((size_t)BS_ * D_);
  uint16_t* ffb   = allocb((size_t)BS_ * FF_);
  uint16_t* pooled= allocb((size_t)B_ * D_);
  uint16_t* z1b   = allocb((size_t)B_ * (D_ / 2));
  uint16_t* in_wb = allocb((size_t)IND_ * D_);
  uint16_t* qkv_wb= allocb((size_t)L_ * D_ * 3 * D_);
  uint16_t* ao_wb = allocb((size_t)L_ * D_ * D_);
  uint16_t* ff_w1b= allocb((size_t)L_ * D_ * FF_);
  uint16_t* ff_w2b= allocb((size_t)L_ * FF_ * D_);
  uint16_t* o_w1b = allocb((size_t)D_ * (D_ / 2));
  uint16_t* o_w2b = allocb((size_t)(D_ / 2) * OUT_);
  // attention logits/probs (bf16, softmax in place): CB batches per pass,
  // chosen deterministically from ws_size.
  size_t used = (size_t)(wp - (char*)d_ws);
  int CB = B_;
  while (CB > 1 && used + (size_t)CB * H_ * S_ * S_ * 2 > ws_size) CB >>= 1;
  uint16_t* attnq = allocb((size_t)CB * H_ * S_ * S_);

  // ---- one-time fp32 -> bf16 conversion of GEMM operand producers ----
  launch_cvt(stream, x,     xb,     (long)BS_ * IND_);
  launch_cvt(stream, in_w,  in_wb,  (long)IND_ * D_);
  launch_cvt(stream, qkv_w, qkv_wb, (long)L_ * D_ * 3 * D_);
  launch_cvt(stream, ao_w,  ao_wb,  (long)L_ * D_ * D_);
  launch_cvt(stream, ff_w1, ff_w1b, (long)L_ * D_ * FF_);
  launch_cvt(stream, ff_w2, ff_w2b, (long)L_ * FF_ * D_);
  launch_cvt(stream, o_w1,  o_w1b,  (long)D_ * (D_ / 2));
  launch_cvt(stream, o_w2,  o_w2b,  (long)(D_ / 2) * OUT_);

  // ---- input projection + LN + pos ----
  launch_gemm(stream, xb, in_wb, in_b, t0, BS_, D_, IND_, IND_, D_, D_,
              1, 1, 0, 0, 0, 0, 0, 0, 1.f, false, false, false);
  ln_kernel<<<BS_, 256, 0, stream>>>(t0, nullptr, pos, in_lng, in_lnb, hres,
                                     hb, D_, S_);

  const float scl = 0.125f;  // 1/sqrt(HD)
  const long sSS = (long)S_ * S_;
  for (int l = 0; l < L_; ++l) {
    launch_gemm(stream, hb, qkv_wb + (size_t)l * D_ * 3 * D_,
                qkv_b + (size_t)l * 3 * D_, qkvb, BS_, 3 * D_, D_, D_, 3 * D_,
                3 * D_, 1, 1, 0, 0, 0, 0, 0, 0, 1.f, false, false, true);
    // attention: all CB batches x H heads per launch triple
    for (int b0 = 0; b0 < B_; b0 += CB) {
      const uint16_t* qb = qkvb + (size_t)b0 * S_ * 3 * D_;
      // logits = scale * Q @ K^T  -> bf16 (z: outer=batch, inner=head)
      launch_gemm(stream, qb, qb + D_, nullptr, attnq, S_, S_, HD_, 3 * D_,
                  3 * D_, S_, CB * H_, H_,
                  HD_, (long)S_ * 3 * D_,           // A: head, batch strides
                  HD_, (long)S_ * 3 * D_,           // B: head, batch strides
                  sSS, (long)H_ * sSS,              // C: head, batch strides
                  scl, true, false, true);
      softmax_kernel<<<CB * H_ * S_, 256, 0, stream>>>(attnq, S_);
      // ctx = probs @ V -> bf16, heads interleaved into [S, D]
      launch_gemm(stream, attnq, qb + 2 * D_, nullptr,
                  ctxb + (size_t)b0 * S_ * D_, S_, HD_, S_, S_, 3 * D_, D_,
                  CB * H_, H_,
                  sSS, (long)H_ * sSS,              // A: head, batch strides
                  HD_, (long)S_ * 3 * D_,           // B: head, batch strides
                  HD_, (long)S_ * D_,               // C: head, batch strides
                  1.f, false, false, true);
    }
    launch_gemm(stream, ctxb, ao_wb + (size_t)l * D_ * D_,
                ao_b + (size_t)l * D_, t0, BS_, D_, D_, D_, D_, D_,
                1, 1, 0, 0, 0, 0, 0, 0, 1.f, false, false, false);
    ln_kernel<<<BS_, 256, 0, stream>>>(hres, t0, nullptr, ln1_g + l * D_,
                                       ln1_b + l * D_, hres, hb, D_, S_);
    launch_gemm(stream, hb, ff_w1b + (size_t)l * D_ * FF_,
                ff_b1 + (size_t)l * FF_, ffb, BS_, FF_, D_, D_, FF_, FF_,
                1, 1, 0, 0, 0, 0, 0, 0, 1.f, false, true, true);
    launch_gemm(stream, ffb, ff_w2b + (size_t)l * FF_ * D_,
                ff_b2 + (size_t)l * D_, t0, BS_, D_, FF_, FF_, D_, D_,
                1, 1, 0, 0, 0, 0, 0, 0, 1.f, false, false, false);
    ln_kernel<<<BS_, 256, 0, stream>>>(hres, t0, nullptr, ln2_g + l * D_,
                                       ln2_b + l * D_, hres, hb, D_, S_);
  }

  pool_kernel<<<B_, 256, 0, stream>>>(hres, pooled, S_, D_);
  launch_gemm(stream, pooled, o_w1b, o_b1, z0, B_, D_ / 2, D_, D_, D_ / 2,
              D_ / 2, 1, 1, 0, 0, 0, 0, 0, 0, 1.f, false, true, false);
  ln_kernel<<<B_, 256, 0, stream>>>(z0, nullptr, nullptr, o_lng, o_lnb, z1f,
                                    z1b, D_ / 2, S_);
  launch_gemm(stream, z1b, o_w2b, o_b2, out, B_, OUT_, D_ / 2, D_ / 2, OUT_,
              OUT_, 1, 1, 0, 0, 0, 0, 0, 0, 1.f, false, false, false);
}
